// TransducerLoss_36232344109737
// MI455X (gfx1250) — compile-verified, matrix operand order checked by reading, not run on hardware
//
#include <hip/hip_runtime.h>
#include <hip/hip_bf16.h>
#include <stdint.h>

typedef __attribute__((ext_vector_type(16))) _Float16 v16h;
typedef __attribute__((ext_vector_type(8)))  float    v8f;
typedef __attribute__((ext_vector_type(4)))  float    v4f;
typedef __attribute__((ext_vector_type(8)))  unsigned int v8u;

#define B_DIM   8
#define T_DIM   256
#define U_DIM   96
#define H_DIM   512
#define NROWS   (B_DIM * T_DIM * U_DIM)     /* 196608 */
#define NGROUPS (NROWS / 16)                /* 12288  */
#define NEGV    (-10000.0f)
#define LOG2E   1.4426950408889634f
#define LN2     0.6931471805599453f

/* LDS row stride (floats): 512 + 4 pad to avoid 16-way bank conflicts */
#define ROWSTRIDE 516
#define BUF_FLOATS (16 * ROWSTRIDE)          /* 8256 floats = 33024 B */
#define WAVES_PER_BLOCK 4
#define LDS_BYTES (WAVES_PER_BLOCK * 2 * BUF_FLOATS * 4)  /* 264192 B <= 320KB WGP */

/* ---- async HBM -> LDS stage: 16 rows x 2KB, b128 per lane, non-temporal ---- */
__device__ __forceinline__ void async_load_group(const float* xg, const float* ldsbuf, int lane)
{
#pragma unroll
  for (int row = 0; row < 16; ++row) {
#pragma unroll
    for (int part = 0; part < 4; ++part) {
      unsigned loff = (unsigned)(uintptr_t)(ldsbuf + row * ROWSTRIDE)
                      + (unsigned)(part * 512 + lane * 16);
      uint64_t ga = (uint64_t)(uintptr_t)xg
                    + (uint64_t)(row * 2048 + part * 512 + lane * 16);
      asm volatile("global_load_async_to_lds_b128 %0, %1, off th:TH_LOAD_NT"
                   :: "v"(loff), "v"(ga) : "memory");
    }
  }
}

__device__ __forceinline__ void wait_async0()
{
  asm volatile("s_wait_asynccnt 0x0" ::: "memory");
}

/* ---------------- Phase 1: log-softmax extraction ---------------- */
__global__ void __launch_bounds__(32 * WAVES_PER_BLOCK)
rnnt_lse_phase1(const float* __restrict__ x, const int* __restrict__ label,
                const int* __restrict__ f_len, const int* __restrict__ y_len,
                const int* __restrict__ pblank, const float* __restrict__ plam,
                float* __restrict__ blank_ws, float* __restrict__ emit_ws)
{
  extern __shared__ float lds[];
  const int lane = threadIdx.x & 31;
  const int wid  = threadIdx.x >> 5;
  const int wave = blockIdx.x * WAVES_PER_BLOCK + wid;
  const int nwaves = gridDim.x * WAVES_PER_BLOCK;
  float* buf0 = lds + (wid * 2 + 0) * BUF_FLOATS;
  float* buf1 = lds + (wid * 2 + 1) * BUF_FLOATS;
  const int   bidx = *pblank;
  const float lam  = *plam;

  v16h bones;
#pragma unroll
  for (int i = 0; i < 16; ++i) bones[i] = (_Float16)1.0f;

  int g = wave;
  if (g >= NGROUPS) return;
  async_load_group(x + (size_t)g * 16 * H_DIM, buf0, lane);
  int d = 0;

  for (; g < NGROUPS; g += nwaves) {
    float* cur = d ? buf1 : buf0;
    float* nxt = d ? buf0 : buf1;
    wait_async0();                               /* cur is resident */
    const int gn = g + nwaves;
    if (gn < NGROUPS)
      async_load_group(x + (size_t)gn * 16 * H_DIM, nxt, lane);  /* overlap */

    const int rowm = lane & 15;

    /* pass 1: row max. lane l -> row (l&15), halves split on lane>=16 */
    {
      const float* rp = cur + rowm * ROWSTRIDE + ((lane < 16) ? 0 : 256);
      float mxl = -1e30f;
#pragma unroll
      for (int i = 0; i < 64; ++i) {
        v4f v = *(const v4f*)(rp + i * 4);
        mxl = fmaxf(mxl, fmaxf(fmaxf(v.x, v.y), fmaxf(v.z, v.w)));
      }
      float mx  = fmaxf(mxl, __shfl_xor(mxl, 16, 32));
      float mxk = mx * LOG2E;   /* exp(x-mx) = exp2(fma(x, log2e, -mxk)) */

      /* pass 2: exp -> packed f16 A tiles, rowsum via WMMA against all-ones B */
      v8f acc = {};
      const float* pr = cur + rowm * ROWSTRIDE + ((lane < 16) ? 0 : 8);
#pragma unroll
      for (int c = 0; c < 16; ++c) {
        const float* q = pr + c * 32;
        v4f x0 = *(const v4f*)(q);
        v4f x1 = *(const v4f*)(q + 4);
        v4f x2 = *(const v4f*)(q + 16);
        v4f x3 = *(const v4f*)(q + 20);
        float xs[16] = { x0.x, x0.y, x0.z, x0.w,  x1.x, x1.y, x1.z, x1.w,
                         x2.x, x2.y, x2.z, x2.w,  x3.x, x3.y, x3.z, x3.w };
        v8u aw;
#pragma unroll
        for (int j = 0; j < 8; ++j) {
          float e0 = __builtin_amdgcn_exp2f(__builtin_fmaf(xs[2 * j + 0], LOG2E, -mxk));
          float e1 = __builtin_amdgcn_exp2f(__builtin_fmaf(xs[2 * j + 1], LOG2E, -mxk));
          /* v_cvt_pk_rtz_f16_f32: two converts per instruction */
          aw[j] = __builtin_bit_cast(unsigned int, __builtin_amdgcn_cvt_pkrtz(e0, e1));
        }
        v16h a = __builtin_bit_cast(v16h, aw);
        acc = __builtin_amdgcn_wmma_f32_16x16x32_f16(
                false, a, false, bones, (short)0, acc, false, false);
      }

      /* extract row m sum: D vgpr (m&7), lanes 0-15 hold M 0..7, 16-31 hold 8..15 */
      float ssel = acc[0];
      const int qsel = lane & 7;
#pragma unroll
      for (int i = 1; i < 8; ++i) ssel = (qsel == i) ? acc[i] : ssel;
      const int m   = lane & 15;
      const int src = (m < 8) ? m : (m + 16);
      float rowsum = __shfl(ssel, src, 32);

      if (lane < 16) {
        const int r  = g * 16 + lane;
        const int u  = r % U_DIM;
        const int bt = r / U_DIM;
        const int t  = bt % T_DIM;
        const int b  = bt / T_DIM;
        const float lse = mx + LN2 * __builtin_amdgcn_logf(rowsum);
        const float* myrow = cur + lane * ROWSTRIDE;
        blank_ws[r] = myrow[bidx] - lse;
        float ev = NEGV;
        if (u < U_DIM - 1 && u < y_len[b]) {
          const int h = label[b * (U_DIM - 1) + u];
          ev = myrow[h] - lse + lam * ((float)(f_len[b] - 1) * 0.5f - (float)t);
        }
        emit_ws[r] = ev;
      }
    }
    d ^= 1;
  }
}

/* ---------------- Phase 2: alpha recursion ---------------- */
__device__ __forceinline__ float laef(float a, float b)
{
  float m = fmaxf(a, b);
  float e = fminf(a, b) - m;      /* <= 0 */
  return m + LN2 * __builtin_amdgcn_logf(1.0f + __builtin_amdgcn_exp2f(e * LOG2E));
}

__global__ void __launch_bounds__(32)
rnnt_alpha_phase2(const float* __restrict__ blank_ws, const float* __restrict__ emit_ws,
                  const int* __restrict__ f_len, const int* __restrict__ y_len,
                  float* __restrict__ out)
{
  const int b    = blockIdx.x;
  const int lane = threadIdx.x;
  const float* eb = emit_ws  + (size_t)b * T_DIM * U_DIM;
  const float* bb = blank_ws + (size_t)b * T_DIM * U_DIM;
  const int fl = f_len[b];
  const int yl = y_len[b];
  const int u0 = 3 * lane;

  /* alpha0 = E row 0 = exclusive cumsum of emit row 0 */
  float em0 = eb[u0], em1 = eb[u0 + 1], em2 = eb[u0 + 2];
  float s0 = em0, s01 = em0 + em1, tot = s01 + em2;
  float inc = tot;
#pragma unroll
  for (int off = 1; off < 32; off <<= 1) {
    float tv = __shfl_up(inc, off, 32);
    if (lane >= off) inc += tv;
  }
  float P = __shfl_up(inc, 1, 32); P = lane ? P : 0.0f;
  float A0 = P, A1 = P + s0, A2 = P + s01;

  /* software-pipelined row loads */
  float nb0 = 0, nb1 = 0, nb2 = 0, ne0 = 0, ne1 = 0, ne2 = 0;
  if (fl > 1) {
    nb0 = bb[u0]; nb1 = bb[u0 + 1]; nb2 = bb[u0 + 2];
    ne0 = eb[U_DIM + u0]; ne1 = eb[U_DIM + u0 + 1]; ne2 = eb[U_DIM + u0 + 2];
  }

  for (int t = 1; t < fl; ++t) {
    const float cb0 = nb0, cb1 = nb1, cb2 = nb2;
    const float ce0 = ne0, ce1 = ne1, ce2 = ne2;
    if (t + 1 < fl) {
      nb0 = bb[t * U_DIM + u0];       nb1 = bb[t * U_DIM + u0 + 1];       nb2 = bb[t * U_DIM + u0 + 2];
      ne0 = eb[(t + 1) * U_DIM + u0]; ne1 = eb[(t + 1) * U_DIM + u0 + 1]; ne2 = eb[(t + 1) * U_DIM + u0 + 2];
    }
    /* E row t: exclusive cumsum of emit row t */
    float t0 = ce0, t01 = ce0 + ce1, tt = t01 + ce2;
    float ic = tt;
#pragma unroll
    for (int off = 1; off < 32; off <<= 1) {
      float tv = __shfl_up(ic, off, 32);
      if (lane >= off) ic += tv;
    }
    float Pp = __shfl_up(ic, 1, 32); Pp = lane ? Pp : 0.0f;
    float E0 = Pp, E1 = Pp + t0, E2 = Pp + t01;

    /* a = alpha_prev + blank[t-1] - E[t]; inclusive log-cumsum-exp over u */
    float q0 = A0 + cb0 - E0, q1 = A1 + cb1 - E1, q2 = A2 + cb2 - E2;
    float i0 = q0, i1 = laef(i0, q1), i2 = laef(i1, q2);
    float sc = i2;
#pragma unroll
    for (int off = 1; off < 32; off <<= 1) {
      float tv = __shfl_up(sc, off, 32);
      if (lane >= off) sc = laef(sc, tv);
    }
    float Pe = __shfl_up(sc, 1, 32); Pe = lane ? Pe : -1e30f;
    A0 = E0 + laef(Pe, i0);
    A1 = E1 + laef(Pe, i1);
    A2 = E2 + laef(Pe, i2);
  }

  const int sl = yl / 3, sk = yl % 3;
  float cand = (sk == 0) ? A0 : (sk == 1) ? A1 : A2;
  float afin = __shfl(cand, sl, 32);
  if (lane == 0) {
    float bfin = bb[(size_t)(fl - 1) * U_DIM + yl];
    out[b] = -(afin + bfin);
  }
}

extern "C" void kernel_launch(void* const* d_in, const int* in_sizes, int n_in,
                              void* d_out, int out_size, void* d_ws, size_t ws_size,
                              hipStream_t stream)
{
  const float* x      = (const float*)d_in[0];
  const int*   label  = (const int*)d_in[1];
  const int*   f_len  = (const int*)d_in[2];
  const int*   y_len  = (const int*)d_in[3];
  const int*   pblank = (const int*)d_in[4];
  const float* plam   = (const float*)d_in[5];
  float*       out    = (float*)d_out;

  float* blank_ws = (float*)d_ws;                 /* B*T*U floats */
  float* emit_ws  = blank_ws + (size_t)NROWS;     /* B*T*U floats */

  rnnt_lse_phase1<<<dim3(256), dim3(32 * WAVES_PER_BLOCK), LDS_BYTES, stream>>>(
      x, label, f_len, y_len, pblank, plam, blank_ws, emit_ws);

  rnnt_alpha_phase2<<<dim3(B_DIM), dim3(32), 0, stream>>>(
      blank_ws, emit_ws, f_len, y_len, out);
}
